// DADecoder_51427938403106
// MI455X (gfx1250) — matrix operand order repaired and checked by reading, not compiled, not run on hardware
//
#include <hip/hip_runtime.h>

// ---------------------------------------------------------------------------
// DADecoder for MI455X (gfx1250, wave32, WMMA).
// Shapes: B=128, S(=T)=128, D=128, H=256.
// All GEMMs use v_wmma_f32_16x16x32_f16 (f16 A/B, f32 accum).
// tanh guaranteed on the TRANS pipe (v_tanh_f32): builtin if declared,
// inline asm otherwise.
// ---------------------------------------------------------------------------

#define NB   128   // batch
#define NS   128   // seq len / timesteps / attention width
#define ND   128   // feature dim
#define NH   256   // hidden
#define KHS  512   // [h,c] concat width
#define KGT  384   // [inp,h] concat width (128 + 256)
#define NGATE 1024 // 4*H

typedef __attribute__((ext_vector_type(8)))  _Float16 v8h;
typedef __attribute__((ext_vector_type(16))) _Float16 v16h;
typedef __attribute__((ext_vector_type(8)))  float    v8f;

__device__ __forceinline__ float fast_tanh(float x) {
#if __has_builtin(__builtin_amdgcn_tanhf)
  return __builtin_amdgcn_tanhf(x);
#elif __has_builtin(__builtin_amdgcn_tanh_f32)
  return __builtin_amdgcn_tanh_f32(x);
#else
  // gfx1250 TRANS-pipe tanh (ISA §8.4 lists TANH among TRANS32 ops).
  float r;
  asm volatile("v_tanh_f32 %0, %1" : "=v"(r) : "v"(x));
  return r;
#endif
}

__device__ __forceinline__ float fast_sigmoid(float x) {
  return 1.0f / (1.0f + __expf(-x));
}

// ---- WMMA fragment loaders (CDNA5 ISA 7.12.2 layouts) ----------------------
// A (16x32 f16, M x K): lane L(0-15) row M=L holds K=[0..7],[16..23];
//                        lane L+16 row M=L holds K=[8..15],[24..31].
__device__ __forceinline__ v16h load_a_frag(const _Float16* __restrict__ A,
                                            int lda, int m0, int k0, int lane) {
  int row = m0 + (lane & 15);
  int kb  = k0 + ((lane >> 4) << 3);          // +0 or +8
  const _Float16* p = A + (size_t)row * lda + kb;
  v8h lo = *(const v8h*)(p);                   // K kb..kb+7
  v8h hi = *(const v8h*)(p + 16);              // K kb+16..kb+23
  return __builtin_shufflevector(lo, hi, 0,1,2,3,4,5,6,7,8,9,10,11,12,13,14,15);
}

// B (32x16 f16, K x N), supplied TRANSPOSED in memory as Bt (N x K row-major):
// lane L(0-15) col N=L holds K=[0..7],[16..23]; lane L+16 col N=L holds K=[8..15],[24..31].
__device__ __forceinline__ v16h load_b_frag(const _Float16* __restrict__ Bt,
                                            int ldb, int n0, int k0, int lane) {
  int col = n0 + (lane & 15);
  int kb  = k0 + ((lane >> 4) << 3);          // +0 or +8
  const _Float16* p = Bt + (size_t)col * ldb + kb;
  v8h lo = *(const v8h*)(p);
  v8h hi = *(const v8h*)(p + 16);
  return __builtin_shufflevector(lo, hi, 0,1,2,3,4,5,6,7,8,9,10,11,12,13,14,15);
}

// C/D (16x16 f32): lane L(0-15) col N=L, VGPR r = row M=r; lanes 16-31 rows 8-15.
__device__ __forceinline__ void store_c_frag(float* __restrict__ C, int ldc,
                                             int m0, int n0, int lane, v8f c) {
  int col = n0 + (lane & 15);
  int rb  = m0 + ((lane >> 4) << 3);
#pragma unroll
  for (int r = 0; r < 8; ++r) C[(size_t)(rb + r) * ldc + col] = c[r];
}

// One 16x16 output tile, K accumulated in steps of 32.
__device__ __forceinline__ void wmma_tile(const _Float16* __restrict__ A, int lda,
                                          const _Float16* __restrict__ Bt, int ldb,
                                          float* __restrict__ C, int ldc,
                                          int m0, int n0, int K, int lane) {
  v8f acc = {};
#pragma unroll 4
  for (int k0 = 0; k0 < K; k0 += 32) {
    v16h a = load_a_frag(A, lda, m0, k0, lane);
    v16h b = load_b_frag(Bt, ldb, n0, k0, lane);
    acc = __builtin_amdgcn_wmma_f32_16x16x32_f16(false, a, false, b,
                                                 (short)0, acc, false, false);
  }
  store_c_frag(C, ldc, m0, n0, lane, acc);
}

// ---- Prep kernels ----------------------------------------------------------

// Convert weights to f16 in the (N,K)-row-major layouts the B-fragments need.
__global__ void dad_prep_weights(const float* __restrict__ W_WU,   // (128, 640)
                                 const float* __restrict__ W_ih,   // (1024,128)
                                 const float* __restrict__ W_hh,   // (1024,256)
                                 const float* __restrict__ b_ih,
                                 const float* __restrict__ b_hh,
                                 _Float16* __restrict__ Wx16,      // (128,128)
                                 _Float16* __restrict__ Wh16,      // (128,512)
                                 _Float16* __restrict__ WcatT,     // (1024,384)
                                 float* __restrict__ bsum) {
  int idx = blockIdx.x * blockDim.x + threadIdx.x;
  if (idx < NGATE * KGT) {
    int o = idx / KGT, k = idx - o * KGT;
    float v = (k < ND) ? W_ih[o * ND + k] : W_hh[o * NH + (k - ND)];
    WcatT[idx] = (_Float16)v;
  }
  if (idx < NS * KHS) {
    int t = idx / KHS, k = idx - t * KHS;
    Wh16[idx] = (_Float16)W_WU[t * (KHS + NS) + k];            // W_h = W_WU[:, :512]
  }
  if (idx < NS * NS) {
    int t = idx / NS, s = idx - t * NS;
    Wx16[idx] = (_Float16)W_WU[t * (KHS + NS) + KHS + s];      // W_x = W_WU[:, 512:]
  }
  if (idx < NGATE) bsum[idx] = b_ih[idx] + b_hh[idx];
}

// Zero initial state: hs16 = [h|c] (128x512), gin = [inp|h] (128x384), c fp32.
__global__ void dad_prep_state(_Float16* __restrict__ hs16,
                               _Float16* __restrict__ gin,
                               float* __restrict__ cbuf) {
  int idx = blockIdx.x * blockDim.x + threadIdx.x;
  if (idx < NB * KHS) hs16[idx] = (_Float16)0.0f;
  if (idx < NB * KGT) gin[idx]  = (_Float16)0.0f;
  if (idx < NB * NH)  cbuf[idx] = 0.0f;
}

// xT16[b,d,s] = (f16) x[b,s,d]  -- B-operand layout for the x_proj GEMMs.
__global__ void dad_transpose_x(const float* __restrict__ x,
                                _Float16* __restrict__ xT16) {
  int idx = blockIdx.x * blockDim.x + threadIdx.x;     // 2M threads
  int b = idx >> 14, r = idx & 16383, d = r >> 7, s = r & 127;
  xT16[idx] = (_Float16)x[(b << 14) + (s << 7) + d];
}

// ---- Phase 1: x_proj[b,t,d] = sum_s W_x[t,s] * x[b,s,d] --------------------
// Per block b: C(128x128) = W_x(128x128) x x_b; A=Wx16, Bt=xT16[b].
__global__ void dad_xproj_gemm(const _Float16* __restrict__ Wx16,
                               const _Float16* __restrict__ xT16,
                               float* __restrict__ xp) {
  int b = blockIdx.x;
  int wave = threadIdx.x >> 5, lane = threadIdx.x & 31;
  const _Float16* Bt = xT16 + (size_t)b * ND * NS;
  float* C = xp + (size_t)b * NS * ND;
  int m0 = wave * 16;                     // 8 waves -> 8 row-tiles
#pragma unroll
  for (int nt = 0; nt < 8; ++nt)
    wmma_tile(Wx16, NS, Bt, NS, C, ND, m0, nt * 16, NS, lane);
}

// ---- Per-step kernels ------------------------------------------------------

// hproj(128x128) = hs16(128x512) x Wh16^T ; K = 512.
__global__ void dad_hproj_gemm(const _Float16* __restrict__ hs16,
                               const _Float16* __restrict__ Wh16,
                               float* __restrict__ hproj) {
  int wave = threadIdx.x >> 5, lane = threadIdx.x & 31;
  int tile = blockIdx.x * 8 + wave;       // 64 tiles (8m x 8n)
  int m0 = (tile >> 3) * 16, n0 = (tile & 7) * 16;
  wmma_tile(hs16, KHS, Wh16, KHS, hproj, NS, m0, n0, KHS, lane);
}

// E -> softmax(d) -> inp ; one block per b, one thread per d.
__global__ void dad_attn_softmax(const float* __restrict__ hproj,
                                 const float* __restrict__ xp,
                                 const float* __restrict__ x,
                                 const float* __restrict__ Wv,
                                 _Float16* __restrict__ gin, int t) {
  __shared__ float sh_h[NS];
  __shared__ float sh_wv[NS];
  __shared__ float rmax[4], rsum[4];
  int b = blockIdx.x, d = threadIdx.x;
  sh_h[d]  = hproj[b * NS + d];
  sh_wv[d] = Wv[d];
  __syncthreads();

  const float* xpb = xp + (size_t)b * NS * ND + d;   // coalesced over d
  float e = 0.0f;
#pragma unroll 4
  for (int s = 0; s < NS; ++s)
    e += sh_wv[s] * fast_tanh(sh_h[s] + xpb[(size_t)s * ND]);

  // block max over 128 threads (4 waves)
  float m = e;
#pragma unroll
  for (int off = 16; off; off >>= 1) m = fmaxf(m, __shfl_xor(m, off, 32));
  int wv = d >> 5;
  if ((d & 31) == 0) rmax[wv] = m;
  __syncthreads();
  m = fmaxf(fmaxf(rmax[0], rmax[1]), fmaxf(rmax[2], rmax[3]));

  float p = __expf(e - m);
  float ps = p;
#pragma unroll
  for (int off = 16; off; off >>= 1) ps += __shfl_xor(ps, off, 32);
  if ((d & 31) == 0) rsum[wv] = ps;
  __syncthreads();
  float sum = rsum[0] + rsum[1] + rsum[2] + rsum[3];

  float alpha = p / sum;
  float inp = x[(size_t)b * NS * ND + (size_t)t * ND + d] * alpha;
  gin[b * KGT + d] = (_Float16)inp;      // A-matrix cols 0..127 for gate GEMM
}

// gates(128x1024) = gin(128x384) x WcatT^T ; K = 384.
__global__ void dad_gates_gemm(const _Float16* __restrict__ gin,
                               const _Float16* __restrict__ WcatT,
                               float* __restrict__ gates) {
  int wave = threadIdx.x >> 5, lane = threadIdx.x & 31;
  int tile = blockIdx.x * 8 + wave;       // 512 tiles (8m x 64n)
  int m0 = (tile >> 6) * 16, n0 = (tile & 63) * 16;
  wmma_tile(gin, KGT, WcatT, KGT, gates, NGATE, m0, n0, KGT, lane);
}

// LSTM pointwise: c/h update, write output + f16 copies for next-step GEMMs.
__global__ void dad_lstm_point(const float* __restrict__ gates,
                               const float* __restrict__ bsum,
                               float* __restrict__ cbuf,
                               _Float16* __restrict__ hs16,
                               _Float16* __restrict__ gin,
                               float* __restrict__ out, int t) {
  int b = blockIdx.x, j = threadIdx.x;    // 256 threads
  const float* g = gates + (size_t)b * NGATE;
  float gi = g[j]           + bsum[j];
  float gf = g[NH + j]      + bsum[NH + j];
  float gg = g[2 * NH + j]  + bsum[2 * NH + j];
  float go = g[3 * NH + j]  + bsum[3 * NH + j];
  float c  = cbuf[b * NH + j];
  float cn = fast_sigmoid(gf) * c + fast_sigmoid(gi) * fast_tanh(gg);
  float hn = fast_sigmoid(go) * fast_tanh(cn);
  cbuf[b * NH + j]          = cn;
  hs16[b * KHS + j]         = (_Float16)hn;   // h half of [h|c]
  hs16[b * KHS + NH + j]    = (_Float16)cn;   // c half of [h|c]
  gin[b * KGT + ND + j]     = (_Float16)hn;   // h cols of [inp|h]
  out[(size_t)t * NB * NH + (size_t)b * NH + j] = hn;  // outs (S,B,H)
}

// ---------------------------------------------------------------------------

extern "C" void kernel_launch(void* const* d_in, const int* in_sizes, int n_in,
                              void* d_out, int out_size, void* d_ws, size_t ws_size,
                              hipStream_t stream) {
  (void)in_sizes; (void)n_in; (void)out_size; (void)ws_size;
  const float* x    = (const float*)d_in[0];
  const float* W_WU = (const float*)d_in[1];
  const float* W_v  = (const float*)d_in[2];
  const float* W_ih = (const float*)d_in[3];
  const float* W_hh = (const float*)d_in[4];
  const float* b_ih = (const float*)d_in[5];
  const float* b_hh = (const float*)d_in[6];
  float* out = (float*)d_out;

  // workspace carve-out (256B aligned)
  char* w = (char*)d_ws;
  auto take = [&](size_t bytes) {
    char* p = w;
    w += (bytes + 255) & ~(size_t)255;
    return (void*)p;
  };
  _Float16* xT16  = (_Float16*)take((size_t)NB * ND * NS * 2);   // 4 MB
  _Float16* Wx16  = (_Float16*)take((size_t)NS * NS * 2);
  _Float16* Wh16  = (_Float16*)take((size_t)NS * KHS * 2);
  _Float16* WcatT = (_Float16*)take((size_t)NGATE * KGT * 2);
  float*    bsum  = (float*)take((size_t)NGATE * 4);
  float*    xp    = (float*)take((size_t)NB * NS * ND * 4);      // 8 MB
  _Float16* hs16  = (_Float16*)take((size_t)NB * KHS * 2);
  _Float16* gin   = (_Float16*)take((size_t)NB * KGT * 2);
  float*    hproj = (float*)take((size_t)NB * NS * 4);
  float*    gates = (float*)take((size_t)NB * NGATE * 4);
  float*    cbuf  = (float*)take((size_t)NB * NH * 4);

  // phase 0: weight conversion + state init + x transpose
  dad_prep_weights<<<(NGATE * KGT + 255) / 256, 256, 0, stream>>>(
      W_WU, W_ih, W_hh, b_ih, b_hh, Wx16, Wh16, WcatT, bsum);
  dad_prep_state<<<(NB * KHS + 255) / 256, 256, 0, stream>>>(hs16, gin, cbuf);
  dad_transpose_x<<<(NB * ND * NS) / 256, 256, 0, stream>>>(x, xT16);

  // phase 1: x_proj (128 independent 128^3 GEMMs)
  dad_xproj_gemm<<<NB, 256, 0, stream>>>(Wx16, xT16, xp);

  // phase 2: sequential scan, 4 kernels per timestep
  for (int t = 0; t < NS; ++t) {
    dad_hproj_gemm<<<8, 256, 0, stream>>>(hs16, Wh16, hproj);
    dad_attn_softmax<<<NB, ND, 0, stream>>>(hproj, xp, x, W_v, gin, t);
    dad_gates_gemm<<<64, 256, 0, stream>>>(gin, WcatT, gates);
    dad_lstm_point<<<NB, NH, 0, stream>>>(gates, bsum, cbuf, hs16, gin, out, t);
  }
}
